// MHAWithPairBias_8899172237441
// MI455X (gfx1250) — matrix operand order repaired and checked by reading, not compiled, not run on hardware
//
#include <hip/hip_runtime.h>
#include <hip/hip_bf16.h>

#define B_ 2
#define S_ 768
#define D_ 128
#define H_ 8
#define C_ 32
#define AH_ 256
#define QT_ (S_/16)          // 48 q-tiles
#define KSPLIT 8
#define KPER (S_/KSPLIT)     // 96 keys per split
#define NCHUNK (KPER/32)     // 3 chunks of 32 keys
#define SCALE 0.35355339059327373f   // H^-0.5 (faithful to reference)
#define LN_EPS 1e-5f

typedef __bf16 bf16;
typedef __attribute__((ext_vector_type(16))) __bf16 v16bf;
typedef __attribute__((ext_vector_type(8)))  __bf16 v8bf;
typedef __attribute__((ext_vector_type(8)))  float  v8f;
typedef __attribute__((ext_vector_type(4)))  float  f4;

static __device__ __forceinline__ v8f wmma_bf16(v16bf a, v16bf b, v8f c) {
  return __builtin_amdgcn_wmma_f32_16x16x32_bf16(false, a, false, b, (short)0, c, false, false);
}

// Build A-fragment (16x32 bf16, K contiguous 16 per lane half) from contiguous fp32 row.
static __device__ __forceinline__ v16bf a_from_f32(const float* rp, int hf, float sc) {
  f4 x0 = *(const f4*)(rp + hf * 16);
  f4 x1 = *(const f4*)(rp + hf * 16 + 4);
  f4 y0 = *(const f4*)(rp + hf * 16 + 8);
  f4 y1 = *(const f4*)(rp + hf * 16 + 12);
  v16bf a;
#pragma unroll
  for (int i = 0; i < 4; ++i) {
    a[i]      = (bf16)(x0[i] * sc);
    a[4 + i]  = (bf16)(x1[i] * sc);
    a[8 + i]  = (bf16)(y0[i] * sc);
    a[12 + i] = (bf16)(y1[i] * sc);
  }
  return a;
}

static __device__ __forceinline__ v16bf cat8(v8bf lo, v8bf hi) {
  v16bf r;
#pragma unroll
  for (int i = 0; i < 8; ++i) { r[i] = lo[i]; r[8 + i] = hi[i]; }
  return r;
}

static __device__ __forceinline__ float rmax16(float v) {
#pragma unroll
  for (int m = 1; m < 16; m <<= 1) v = fmaxf(v, __shfl_xor(v, m, 32));
  return v;
}
static __device__ __forceinline__ float rsum16(float v) {
#pragma unroll
  for (int m = 1; m < 16; m <<= 1) v += __shfl_xor(v, m, 32);
  return v;
}

// ---------------- kernel 0: transpose + bf16-convert weights (tiny) ----------------
__global__ void prep_weights(const float* __restrict__ Wq, const float* __restrict__ Wk,
                             const float* __restrict__ Wv, const float* __restrict__ Wb,
                             const float* __restrict__ Wo,
                             bf16* __restrict__ WqT, bf16* __restrict__ WkT,
                             bf16* __restrict__ WvT, bf16* __restrict__ WbT,
                             bf16* __restrict__ WoT) {
  int t = threadIdx.x;
  for (int i = t; i < AH_ * D_; i += 256) {       // W[D][AH] -> WT[AH][D]
    int n = i / D_, d = i % D_;
    WqT[i] = (bf16)Wq[d * AH_ + n];
    WkT[i] = (bf16)Wk[d * AH_ + n];
    WvT[i] = (bf16)Wv[d * AH_ + n];
  }
  for (int i = t; i < D_ * AH_; i += 256) {       // Wo[AH][D] -> WoT[D][AH]
    int n = i / AH_, kk = i % AH_;
    WoT[i] = (bf16)Wo[kk * D_ + n];
  }
  for (int i = t; i < 16 * D_; i += 256) {        // Wb[D][H] -> WbT[16][D], zero-padded
    int n = i / D_, d = i % D_;
    WbT[i] = (n < H_) ? (bf16)Wb[d * H_ + n] : (bf16)0.0f;
  }
}

// ---------------- kernel 1: QKV projections via WMMA ----------------
__global__ void qkv_proj(const float* __restrict__ q, const float* __restrict__ k,
                         const float* __restrict__ v,
                         const bf16* __restrict__ WqT, const bf16* __restrict__ WkT,
                         const bf16* __restrict__ WvT,
                         bf16* __restrict__ qh, bf16* __restrict__ kh,
                         bf16* __restrict__ vhT) {
  const int tid = threadIdx.x, lane = tid & 31, wave = tid >> 5;
  const int ln = lane & 15, hf = lane >> 4;
  const int b = blockIdx.x / QT_, s0 = (blockIdx.x % QT_) * 16;

#pragma unroll
  for (int i = 0; i < 6; ++i) {
    int job = wave * 6 + i;            // 48 jobs: 3 matrices x 16 N-tiles
    int mat = job >> 4, nt = job & 15;
    const float* x  = (mat == 0) ? q : (mat == 1) ? k : v;
    const bf16*  wt = (mat == 0) ? WqT : (mat == 1) ? WkT : WvT;
    float sc = (mat == 0) ? SCALE : 1.0f;      // fold H^-0.5 into qh
    const float* rowp = x + ((size_t)(b * S_ + s0 + ln)) * D_;
    v8f acc = {};
#pragma unroll
    for (int kc = 0; kc < 4; ++kc) {
      v16bf a  = a_from_f32(rowp + kc * 32, hf, sc);
      v16bf wb = *(const v16bf*)&wt[(size_t)(nt * 16 + ln) * D_ + kc * 32 + hf * 16];
      acc = wmma_bf16(a, wb, acc);
    }
    if (mat < 2) {
      bf16* dst = (mat == 0) ? qh : kh;
#pragma unroll
      for (int r = 0; r < 8; ++r)
        dst[((size_t)(b * S_ + s0 + r + 8 * hf)) * AH_ + nt * 16 + ln] = (bf16)acc[r];
    } else {  // vh stored transposed: vhT[b*AH + c][s] so ctx B-fragments are contiguous
#pragma unroll
      for (int r = 0; r < 8; ++r)
        vhT[((size_t)(b * AH_ + nt * 16 + ln)) * S_ + s0 + r + 8 * hf] = (bf16)acc[r];
    }
  }
}

// ---------------- kernel 2: fused pair-bias + flash attention (K-split partials) ----------------
__global__ void attn_partial(const bf16* __restrict__ qh, const bf16* __restrict__ kh,
                             const bf16* __restrict__ vhT, const bf16* __restrict__ WbT,
                             const float* __restrict__ bias, const int* __restrict__ mask,
                             float* __restrict__ pctx, float* __restrict__ pml) {
  __shared__ __align__(16) float pb[512][9];        // 512 (q,k) pairs x 8 heads (+pad)
  __shared__ __align__(32) bf16  pl[8][16][32];     // per-wave P tile for lane reshuffle

  const int tid = threadIdx.x, lane = tid & 31, wave = tid >> 5;
  const int ln = lane & 15, hf = lane >> 4;
  const int blk = blockIdx.x;
  const int ks = blk & (KSPLIT - 1);
  const int qt = (blk >> 3) % QT_;
  const int b  = blk / (KSPLIT * QT_);
  const int q0 = qt * 16;
  const int h  = wave;                              // one head per wave

  // Wb B-fragments for the 4 K-chunks of D=128 (kept in registers)
  v16bf wbf[4];
#pragma unroll
  for (int kc = 0; kc < 4; ++kc)
    wbf[kc] = *(const v16bf*)&WbT[(size_t)ln * D_ + kc * 32 + hf * 16];

  // q A-fragment for this head (16 rows x 32 c)
  const bf16* qrow = qh + ((size_t)(b * S_ + q0 + ln)) * AH_ + h * C_;
  v16bf qa = cat8(*(const v8bf*)(qrow + hf * 16), *(const v8bf*)(qrow + hf * 16 + 8));

  float mrun[8], lrun[8];
  v8f acc0 = {}, acc1 = {};
#pragma unroll
  for (int r = 0; r < 8; ++r) { mrun[r] = -1e30f; lrun[r] = 0.0f; }

  for (int ch = 0; ch < NCHUNK; ++ch) {
    const int k0 = ks * KPER + ch * 32;

    // ---- phase A: cooperative pair-bias (bias @ Wb) for 16q x 32k pairs ----
#pragma unroll
    for (int mt = 0; mt < 4; ++mt) {
      const int mtile = wave * 4 + mt;
      const int pair  = mtile * 16 + ln;            // A-matrix row
      const int ql = pair >> 5, kl = pair & 31;
      const float* bp = bias + ((size_t)(b * S_ + q0 + ql) * S_ + (k0 + kl)) * D_;
      if (ch + 1 < NCHUNK) __builtin_prefetch(bp + 32 * D_, 0, 0);  // next k-chunk
      v8f acc = {};
#pragma unroll
      for (int kc = 0; kc < 4; ++kc) {
        v16bf a = a_from_f32(bp + kc * 32, hf, 1.0f);
        acc = wmma_bf16(a, wbf[kc], acc);
      }
      if (ln < H_) {                                // cols 8..15 are padding
#pragma unroll
        for (int r = 0; r < 8; ++r) {
          int p = mtile * 16 + r + 8 * hf;
          int mq = p >> 5, mk = p & 31;
          int mv = mask[(size_t)(b * S_ + q0 + mq) * S_ + (k0 + mk)];
          pb[p][ln] = acc[r] + ((float)mv * 1.0e9f - 1.0f);   // fold mask term
        }
      }
    }
    __syncthreads();

    // ---- phase B: per-head logits, online softmax, ctx accumulation ----
    const bf16* kr0 = kh + ((size_t)(b * S_ + k0 + ln)) * AH_ + h * C_ + hf * 16;
    const bf16* kr1 = kh + ((size_t)(b * S_ + k0 + 16 + ln)) * AH_ + h * C_ + hf * 16;
    v8f z = {};
    v8f L0 = wmma_bf16(qa, *(const v16bf*)kr0, z);
    v8f L1 = wmma_bf16(qa, *(const v16bf*)kr1, z);
#pragma unroll
    for (int r = 0; r < 8; ++r) {
      int m = r + 8 * hf;
      L0[r] += pb[m * 32 + ln][h];
      L1[r] += pb[m * 32 + 16 + ln][h];
    }
#pragma unroll
    for (int r = 0; r < 8; ++r) {
      int m = r + 8 * hf;
      float t  = rmax16(fmaxf(L0[r], L1[r]));
      float mn = fmaxf(mrun[r], t);
      float sc = __expf(mrun[r] - mn);
      float p0 = __expf(L0[r] - mn);
      float p1 = __expf(L1[r] - mn);
      lrun[r] = lrun[r] * sc + rsum16(p0 + p1);
      mrun[r] = mn;
      acc0[r] *= sc; acc1[r] *= sc;
      pl[wave][m][ln]      = (bf16)p0;
      pl[wave][m][16 + ln] = (bf16)p1;
    }
    // same-wave LDS round-trip to reshuffle P into A-fragment layout
    asm volatile("s_wait_dscnt 0" ::: "memory");
    __builtin_amdgcn_wave_barrier();
    v16bf pa = cat8(*(const v8bf*)&pl[wave][ln][hf * 16],
                    *(const v8bf*)&pl[wave][ln][hf * 16 + 8]);
    const bf16* v0 = vhT + ((size_t)(b * AH_ + h * C_ + ln)) * S_ + k0 + hf * 16;
    const bf16* v1 = vhT + ((size_t)(b * AH_ + h * C_ + 16 + ln)) * S_ + k0 + hf * 16;
    acc0 = wmma_bf16(pa, *(const v16bf*)v0, acc0);
    acc1 = wmma_bf16(pa, *(const v16bf*)v1, acc1);
    __syncthreads();   // pb is rewritten next chunk
  }

  // ---- write flash partials ----
  size_t base = ((size_t)(b * QT_ + qt) * KSPLIT + ks) * H_ + h;
  float* cp = pctx + base * 16 * 32;
#pragma unroll
  for (int r = 0; r < 8; ++r) {
    int m = r + 8 * hf;
    cp[m * 32 + ln]      = acc0[r];
    cp[m * 32 + 16 + ln] = acc1[r];
  }
  if (ln == 0) {
    float* mp = pml + base * 32;
#pragma unroll
    for (int r = 0; r < 8; ++r) {
      int m = r + 8 * hf;
      mp[m * 2]     = mrun[r];
      mp[m * 2 + 1] = lrun[r];
    }
  }
}

// ---------------- kernel 3: combine K-splits, output projection (WMMA), LayerNorm ----------------
__global__ void combine_out(const float* __restrict__ pctx, const float* __restrict__ pml,
                            const bf16* __restrict__ WoT, const float* __restrict__ bo,
                            const float* __restrict__ ln_g, const float* __restrict__ ln_b,
                            float* __restrict__ out) {
  __shared__ __align__(32) bf16  cx[16][AH_];
  __shared__ float ol[16][132];
  __shared__ float muS[16], rsS[16];

  const int tid = threadIdx.x, lane = tid & 31, wave = tid >> 5;
  const int ln = lane & 15, hf = lane >> 4;
  const int qt = blockIdx.x % QT_, b = blockIdx.x / QT_, q0 = qt * 16;
  const int h = wave;

#pragma unroll
  for (int r = 0; r < 8; ++r) {
    int m = r + 8 * hf;
    float mt = -1e30f;
    for (int ks = 0; ks < KSPLIT; ++ks) {
      size_t bs = ((size_t)(b * QT_ + qt) * KSPLIT + ks) * H_ + h;
      mt = fmaxf(mt, pml[bs * 32 + m * 2]);
    }
    float lt = 0.f, c0 = 0.f, c1 = 0.f;
    for (int ks = 0; ks < KSPLIT; ++ks) {
      size_t bs = ((size_t)(b * QT_ + qt) * KSPLIT + ks) * H_ + h;
      float w = __expf(pml[bs * 32 + m * 2] - mt);
      lt += w * pml[bs * 32 + m * 2 + 1];
      const float* cp = pctx + (bs * 16 + m) * 32;
      c0 += w * cp[ln];
      c1 += w * cp[16 + ln];
    }
    float inv = 1.0f / lt;
    cx[m][h * C_ + ln]      = (bf16)(c0 * inv);
    cx[m][h * C_ + 16 + ln] = (bf16)(c1 * inv);
  }
  __syncthreads();

  // output projection: ctx(16x256) @ Wo(256x128); wave = N-tile (16 output cols)
  v8f acc = {};
#pragma unroll
  for (int kc = 0; kc < 8; ++kc) {
    v16bf a  = cat8(*(const v8bf*)&cx[ln][kc * 32 + hf * 16],
                    *(const v8bf*)&cx[ln][kc * 32 + hf * 16 + 8]);
    v16bf wb = *(const v16bf*)&WoT[(size_t)(wave * 16 + ln) * AH_ + kc * 32 + hf * 16];
    acc = wmma_bf16(a, wb, acc);
  }
  int col = wave * 16 + ln;
  float bov = bo[col];
#pragma unroll
  for (int r = 0; r < 8; ++r) ol[r + 8 * hf][col] = acc[r] + bov;
  __syncthreads();

  if (tid < 16) {
    float s = 0.f, s2 = 0.f;
    for (int j = 0; j < D_; ++j) { float x = ol[tid][j]; s += x; s2 += x * x; }
    float mu = s / D_;
    muS[tid] = mu;
    rsS[tid] = rsqrtf(s2 / D_ - mu * mu + LN_EPS);
  }
  __syncthreads();
  int row = tid >> 4, cb = (tid & 15) * 8;
#pragma unroll
  for (int i = 0; i < 8; ++i) {
    int cc = cb + i;
    float x = (ol[row][cc] - muS[row]) * rsS[row] * ln_g[cc] + ln_b[cc];
    out[((size_t)(b * S_ + q0 + row)) * D_ + cc] = x;
  }
}

extern "C" void kernel_launch(void* const* d_in, const int* in_sizes, int n_in,
                              void* d_out, int out_size, void* d_ws, size_t ws_size,
                              hipStream_t stream) {
  (void)in_sizes; (void)n_in; (void)out_size; (void)ws_size;
  const float* q    = (const float*)d_in[0];
  const float* k    = (const float*)d_in[1];
  const float* v    = (const float*)d_in[2];
  const int*   mask = (const int*)  d_in[3];
  const float* bias = (const float*)d_in[4];
  const float* Wq   = (const float*)d_in[5];
  const float* Wk   = (const float*)d_in[6];
  const float* Wv   = (const float*)d_in[7];
  const float* Wb   = (const float*)d_in[8];
  const float* Wo   = (const float*)d_in[9];
  const float* bo   = (const float*)d_in[10];
  const float* lg   = (const float*)d_in[11];
  const float* lb   = (const float*)d_in[12];
  float* out = (float*)d_out;

  size_t off = 0;
  auto al = [&](size_t bytes) { size_t r = off; off += (bytes + 255) & ~(size_t)255; return r; };
  char* ws = (char*)d_ws;
  bf16* WqT = (bf16*)(ws + al((size_t)AH_ * D_ * 2));
  bf16* WkT = (bf16*)(ws + al((size_t)AH_ * D_ * 2));
  bf16* WvT = (bf16*)(ws + al((size_t)AH_ * D_ * 2));
  bf16* WbT = (bf16*)(ws + al((size_t)16 * D_ * 2));
  bf16* WoT = (bf16*)(ws + al((size_t)D_ * AH_ * 2));
  bf16* qh  = (bf16*)(ws + al((size_t)B_ * S_ * AH_ * 2));
  bf16* kh  = (bf16*)(ws + al((size_t)B_ * S_ * AH_ * 2));
  bf16* vhT = (bf16*)(ws + al((size_t)B_ * S_ * AH_ * 2));
  float* pctx = (float*)(ws + al((size_t)B_ * QT_ * KSPLIT * H_ * 16 * 32 * 4));
  float* pml  = (float*)(ws + al((size_t)B_ * QT_ * KSPLIT * H_ * 32 * 4));

  prep_weights<<<1, 256, 0, stream>>>(Wq, Wk, Wv, Wb, Wo, WqT, WkT, WvT, WbT, WoT);
  qkv_proj<<<B_ * QT_, 256, 0, stream>>>(q, k, v, WqT, WkT, WvT, qh, kh, vhT);
  attn_partial<<<B_ * QT_ * KSPLIT, 256, 0, stream>>>(qh, kh, vhT, WbT, bias, mask, pctx, pml);
  combine_out<<<B_ * QT_, 256, 0, stream>>>(pctx, pml, WoT, bo, lg, lb, out);
}
